// SimpleLSTM_70377334112761
// MI455X (gfx1250) — matrix operand order repaired and checked by reading, not compiled, not run on hardware
//
#include <hip/hip_runtime.h>
#include <hip/hip_bf16.h>
#include <math.h>

// MI455X / gfx1250, wave32. WMMA f16 16x16x32, f32 accumulate.
//
//  K1 pack_b_frags : repack W (f32 [K,512]) -> f16 B-fragments in WMMA lane layout
//  K2 input_gemm   : xz1[t,b,:] = x[b,t,:] @ W1 + b1   (parallel over T*B, f16 out)
//  K3 lstm_fused   : per-workgroup 32-batch-row tile runs both LSTM layers for all
//                    256 steps fused; h kept in LDS *in f16 A-fragment layout* so
//                    each GEMM A-load is one 32B/lane ds_load_b128 pair with no
//                    f32->f16 conversion on the sequential critical path.

typedef __attribute__((ext_vector_type(16))) _Float16 v16h;
typedef __attribute__((ext_vector_type(8)))  float    v8f;
typedef __attribute__((ext_vector_type(4)))  float    v4f;

#define NCOL 512          // 4*U
#define NT_TOTAL 32       // 512/16 n-tiles
#define T_STEPS 256
#define BATCH 512
#define BT 32             // batch rows per workgroup in recurrence
#define H_STRIDE 132      // padded LDS stride for c (banks rotate: 132%64=4)
#define Z_STRIDE 516      // padded LDS stride for z (rows m,m+8 differ by 32 banks)

__device__ __forceinline__ float fast_sigmoid(float x) {
  return 1.f / (1.f + __expf(-x));
}
__device__ __forceinline__ float fast_tanh(float x) {
  float t = __expf(-2.f * x);          // v_exp_f32
  return (1.f - t) / (1.f + t);
}

// A-fragment slot for element (m,k), m in [0,32), k in [0,128):
//   mt = m>>4, lane = ((k>>3)&1)*16 + (m&15), kb = k>>5,
//   j  = (k&7) + 8*((k>>4)&1);  flat = ((mt*4+kb)*32+lane)*16 + j
__device__ __forceinline__ int afrag_off(int m, int k) {
  int mt   = m >> 4;
  int lane = (((k >> 3) & 1) << 4) + (m & 15);
  int kb   = k >> 5;
  int j    = (k & 7) + ((k >> 4) & 1) * 8;
  return ((mt * 4 + kb) * 32 + lane) * 16 + j;
}

// ---------------------------------------------------------------------------
// K1: pack a [K,512] f32 row-major matrix into f16 WMMA B fragments.
// Fragment (kb,nt): lane l -> n = nt*16 + (l&15), k = kb*32 + (j>>3)*16 + (l>>4)*8 + (j&7).
// ---------------------------------------------------------------------------
__global__ void pack_b_frags(const float* __restrict__ W, _Float16* __restrict__ out,
                             int K) {
  int idx = blockIdx.x * blockDim.x + threadIdx.x;
  int total = K * NCOL;
  if (idx >= total) return;
  int j    = idx & 15;
  int lane = (idx >> 4) & 31;
  int frag = idx >> 9;               // kb*NT_TOTAL + nt
  int kb = frag / NT_TOTAL;
  int nt = frag % NT_TOTAL;
  int n  = nt * 16 + (lane & 15);
  int k  = kb * 32 + ((j >> 3) << 4) + ((lane >> 4) << 3) + (j & 7);
  out[idx] = (_Float16)W[(size_t)k * NCOL + n];
}

// ---------------------------------------------------------------------------
// K2: xz1 = x_tm @ W1 + b1, time-major rows r = t*B + b (x is [B,T,F]).
// One workgroup = one 16-row m-tile, 8 waves x 4 n-tiles each = full N=512.
// ---------------------------------------------------------------------------
__global__ __launch_bounds__(256, 1)
void input_gemm(const float* __restrict__ x, const _Float16* __restrict__ pW1,
                const float* __restrict__ b1, _Float16* __restrict__ xz1) {
  __shared__ float xs[16][68];       // 16 rows x 64 cols, padded

  const int tid  = threadIdx.x;
  const int lane = tid & 31;
  const int w    = tid >> 5;         // wave 0..7
  const int rowbase = blockIdx.x * 16;

  // stage 16 rows of x (row R=(t,b) -> x[b, t, 0:64]) into LDS
  {
    int r  = tid >> 4;               // 0..15
    int c4 = (tid & 15) * 4;         // 0,4,..,60
    int R  = rowbase + r;
    int t  = R / BATCH, b = R % BATCH;
    const float* src = x + ((size_t)b * T_STEPS + t) * 64 + c4;
    v4f v = *(const v4f*)src;
    xs[r][c4 + 0] = v[0]; xs[r][c4 + 1] = v[1];
    xs[r][c4 + 2] = v[2]; xs[r][c4 + 3] = v[3];
  }
  __syncthreads();

  const int m16   = lane & 15;
  const int abase = (lane >> 4) * 8;

  v8f acc[4];
  #pragma unroll
  for (int i = 0; i < 4; ++i) acc[i] = {};

  #pragma unroll
  for (int kb = 0; kb < 2; ++kb) {
    v16h a;
    const float* hp = &xs[m16][kb * 32];
    #pragma unroll
    for (int j = 0; j < 8; ++j) a[j]     = (_Float16)hp[abase + j];
    #pragma unroll
    for (int j = 0; j < 8; ++j) a[8 + j] = (_Float16)hp[16 + abase + j];
    #pragma unroll
    for (int n = 0; n < 4; ++n) {
      int nt = w * 4 + n;
      v16h bf = *(const v16h*)(pW1 + (((size_t)kb * NT_TOTAL + nt) * 32 + lane) * 16);
      acc[n] = __builtin_amdgcn_wmma_f32_16x16x32_f16(false, a, false, bf,
                                                      (short)0, acc[n], false, false);
    }
  }

  #pragma unroll
  for (int n = 0; n < 4; ++n) {
    int nt  = w * 4 + n;
    int col = nt * 16 + (lane & 15);
    float bias = b1[col];
    #pragma unroll
    for (int j = 0; j < 8; ++j) {
      int m = j + 8 * (lane >> 4);
      size_t R = (size_t)(rowbase + m);
      xz1[R * NCOL + col] = (_Float16)(acc[n][j] + bias);
    }
  }
}

// ---------------------------------------------------------------------------
// K3: fused 2-layer LSTM recurrence + Dense. 16 workgroups x 32 batch rows.
// 8 waves: wave w -> m-tile (w>>2), n-slice (w&3) of 8 n-tiles.
// ---------------------------------------------------------------------------
__global__ __launch_bounds__(256, 1)
void lstm_fused(const _Float16* __restrict__ xz1,
                const _Float16* __restrict__ pU1,
                const _Float16* __restrict__ pW2,
                const _Float16* __restrict__ pU2,
                const float* __restrict__ b2,
                const float* __restrict__ Wd,
                const float* __restrict__ bd,
                float* __restrict__ out) {
  // h kept as f16 WMMA A-fragments: 2 mtiles * 4 kb * 32 lanes * 16 halves = 8KB
  __shared__ alignas(64) _Float16 hA1[2 * 4 * 32 * 16];
  __shared__ alignas(64) _Float16 hA2[2 * 4 * 32 * 16];
  __shared__ float c1s[BT][H_STRIDE];
  __shared__ float c2s[BT][H_STRIDE];
  __shared__ float zb[BT][Z_STRIDE];
  __shared__ float b2s[NCOL];

  const int tid  = threadIdx.x;
  const int lane = tid & 31;
  const int w    = tid >> 5;
  const int mt   = w >> 2;                   // 0..1
  const int ns   = w & 3;                    // n-slice: nt = ns*8 .. ns*8+7
  const int b0   = blockIdx.x * BT;

  for (int i = tid; i < 2 * 4 * 32 * 16; i += 256) {
    hA1[i] = (_Float16)0.f;
    hA2[i] = (_Float16)0.f;
  }
  for (int i = tid; i < BT * H_STRIDE; i += 256) {
    ((float*)c1s)[i] = 0.f;
    ((float*)c2s)[i] = 0.f;
  }
  for (int i = tid; i < NCOL; i += 256) b2s[i] = b2[i];
  __syncthreads();

  for (int t = 0; t < T_STEPS; ++t) {
    const _Float16* xzt = xz1 + ((size_t)t * BATCH + b0) * NCOL;

    // ======== layer 1 GEMM: z = h1 @ U1 ========
    {
      v8f acc[8];
      #pragma unroll
      for (int i = 0; i < 8; ++i) acc[i] = {};
      #pragma unroll
      for (int kb = 0; kb < 4; ++kb) {
        v16h a = *(const v16h*)(hA1 + ((mt * 4 + kb) * 32 + lane) * 16);
        #pragma unroll
        for (int n = 0; n < 8; ++n) {
          int nt = ns * 8 + n;
          v16h bf = *(const v16h*)(pU1 + (((size_t)kb * NT_TOTAL + nt) * 32 + lane) * 16);
          acc[n] = __builtin_amdgcn_wmma_f32_16x16x32_f16(false, a, false, bf,
                                                          (short)0, acc[n], false, false);
        }
      }
      #pragma unroll
      for (int n = 0; n < 8; ++n) {
        int col = (ns * 8 + n) * 16 + (lane & 15);
        #pragma unroll
        for (int j = 0; j < 8; ++j) {
          int m = mt * 16 + j + 8 * (lane >> 4);
          zb[m][col] = acc[n][j];
        }
      }
    }
    __syncthreads();

    // ======== layer 1 gates (xz1 added here, coalesced f16 reads) ========
    for (int e = tid; e < BT * 128; e += 256) {
      int b = e >> 7, u = e & 127;
      const _Float16* xr = xzt + (size_t)b * NCOL;
      float zi = zb[b][u]       + (float)xr[u];
      float zf = zb[b][128 + u] + (float)xr[128 + u];
      float zg = zb[b][256 + u] + (float)xr[256 + u];
      float zo = zb[b][384 + u] + (float)xr[384 + u];
      float ig = fast_sigmoid(zi);
      float fg = fast_sigmoid(zf);
      float gg = fast_tanh(zg);
      float og = fast_sigmoid(zo);
      float c  = fg * c1s[b][u] + ig * gg;
      c1s[b][u] = c;
      hA1[afrag_off(b, u)] = (_Float16)(og * fast_tanh(c));
    }
    // prefetch next step's 32KB xz tile while gates run (global_prefetch_b8)
    if (t + 1 < T_STEPS) {
      const char* nxt = (const char*)(xz1 + ((size_t)(t + 1) * BATCH + b0) * NCOL);
      __builtin_prefetch(nxt + (size_t)tid * 128, 0, 1);
    }
    __syncthreads();

    // ======== layer 2 GEMM: z = h1 @ W2 + h2 @ U2 ========
    {
      v8f acc[8];
      #pragma unroll
      for (int i = 0; i < 8; ++i) acc[i] = {};
      #pragma unroll
      for (int kb = 0; kb < 4; ++kb) {
        v16h a = *(const v16h*)(hA1 + ((mt * 4 + kb) * 32 + lane) * 16);
        #pragma unroll
        for (int n = 0; n < 8; ++n) {
          int nt = ns * 8 + n;
          v16h bf = *(const v16h*)(pW2 + (((size_t)kb * NT_TOTAL + nt) * 32 + lane) * 16);
          acc[n] = __builtin_amdgcn_wmma_f32_16x16x32_f16(false, a, false, bf,
                                                          (short)0, acc[n], false, false);
        }
      }
      #pragma unroll
      for (int kb = 0; kb < 4; ++kb) {
        v16h a = *(const v16h*)(hA2 + ((mt * 4 + kb) * 32 + lane) * 16);
        #pragma unroll
        for (int n = 0; n < 8; ++n) {
          int nt = ns * 8 + n;
          v16h bf = *(const v16h*)(pU2 + (((size_t)kb * NT_TOTAL + nt) * 32 + lane) * 16);
          acc[n] = __builtin_amdgcn_wmma_f32_16x16x32_f16(false, a, false, bf,
                                                          (short)0, acc[n], false, false);
        }
      }
      #pragma unroll
      for (int n = 0; n < 8; ++n) {
        int col = (ns * 8 + n) * 16 + (lane & 15);
        #pragma unroll
        for (int j = 0; j < 8; ++j) {
          int m = mt * 16 + j + 8 * (lane >> 4);
          zb[m][col] = acc[n][j];
        }
      }
    }
    __syncthreads();

    // ======== layer 2 gates (b2 added here) ========
    for (int e = tid; e < BT * 128; e += 256) {
      int b = e >> 7, u = e & 127;
      float zi = zb[b][u]       + b2s[u];
      float zf = zb[b][128 + u] + b2s[128 + u];
      float zg = zb[b][256 + u] + b2s[256 + u];
      float zo = zb[b][384 + u] + b2s[384 + u];
      float ig = fast_sigmoid(zi);
      float fg = fast_sigmoid(zf);
      float gg = fast_tanh(zg);
      float og = fast_sigmoid(zo);
      float c  = fg * c2s[b][u] + ig * gg;
      c2s[b][u] = c;
      hA2[afrag_off(b, u)] = (_Float16)(og * fast_tanh(c));
    }
    __syncthreads();
  }

  // ======== Dense(1, relu) on final h2 (read from fragment layout) ========
  if (tid < BT) {
    float s = 0.f;
    #pragma unroll 4
    for (int u = 0; u < 128; ++u) s += (float)hA2[afrag_off(tid, u)] * Wd[u];
    s += bd[0];
    out[b0 + tid] = s > 0.f ? s : 0.f;
  }
}

// ---------------------------------------------------------------------------
extern "C" void kernel_launch(void* const* d_in, const int* in_sizes, int n_in,
                              void* d_out, int out_size, void* d_ws, size_t ws_size,
                              hipStream_t stream) {
  const float* x  = (const float*)d_in[0];   // [512,256,64]
  const float* W1 = (const float*)d_in[1];   // [64,512]
  const float* U1 = (const float*)d_in[2];   // [128,512]
  const float* b1 = (const float*)d_in[3];   // [512]
  const float* W2 = (const float*)d_in[4];   // [128,512]
  const float* U2 = (const float*)d_in[5];   // [128,512]
  const float* b2 = (const float*)d_in[6];   // [512]
  const float* Wd = (const float*)d_in[7];   // [128]
  const float* bd = (const float*)d_in[8];   // [1]
  float* out = (float*)d_out;                // [512]

  // workspace carve (all 256B-aligned)
  char* ws = (char*)d_ws;
  _Float16* pW1 = (_Float16*)(ws);                               // 64KB
  _Float16* pU1 = (_Float16*)(ws + 65536);                       // 128KB
  _Float16* pW2 = (_Float16*)(ws + 65536 + 131072);
  _Float16* pU2 = (_Float16*)(ws + 65536 + 2 * 131072);
  _Float16* xz1 = (_Float16*)(ws + 65536 + 3 * 131072);          // 128MB

  pack_b_frags<<<(64 * NCOL + 255) / 256, 256, 0, stream>>>(W1, pW1, 64);
  pack_b_frags<<<(128 * NCOL + 255) / 256, 256, 0, stream>>>(U1, pU1, 128);
  pack_b_frags<<<(128 * NCOL + 255) / 256, 256, 0, stream>>>(W2, pW2, 128);
  pack_b_frags<<<(128 * NCOL + 255) / 256, 256, 0, stream>>>(U2, pU2, 128);

  input_gemm<<<(T_STEPS * BATCH) / 16, 256, 0, stream>>>(x, pW1, b1, xz1);

  lstm_fused<<<BATCH / BT, 256, 0, stream>>>(xz1, pU1, pW2, pU2, b2, Wd, bd, out);
}
